// RNNGraphConvModule_86260123174010
// MI455X (gfx1250) — matrix operand order repaired and checked by reading, not compiled, not run on hardware
//
#include <hip/hip_runtime.h>

#define N_NODES 50000
#define E_EDGES 800000
#define NCF 16
#define FDIM 13
#define HID 32

#define NTILES (E_EDGES / 16)   // 50000 edge tiles of 16
#define FBLOCKS 625             // 625 blocks * 8 waves = 5000 waves
#define NWAVES (FBLOCKS * 8)
#define TPW (NTILES / NWAVES)   // 10 tiles per wave, exact

typedef __attribute__((ext_vector_type(16))) _Float16 v16h;
typedef __attribute__((ext_vector_type(8)))  _Float16 v8h;
typedef __attribute__((ext_vector_type(8)))  float    v8f;

// ---------------------------------------------------------------------------
// Zero-fill helper (agg / deg must be zeroed; harness poisons ws with 0xAA).
// ---------------------------------------------------------------------------
__global__ __launch_bounds__(256) void fill_zero_kernel(float* __restrict__ p, int n) {
    int i = blockIdx.x * 256 + threadIdx.x;
    if (i < n) p[i] = 0.0f;
}

// ---------------------------------------------------------------------------
// Degree computation: deg[dst] += 1  (float atomic, exact for small counts)
// ---------------------------------------------------------------------------
__global__ __launch_bounds__(256) void deg_kernel(const int* __restrict__ dst,
                                                  float* __restrict__ deg) {
    int e = blockIdx.x * 256 + threadIdx.x;
    if (e < E_EDGES) unsafeAtomicAdd(&deg[dst[e]], 1.0f);
}

// ---------------------------------------------------------------------------
// Filter-generating network, fully WMMA:
//   layer 1:  h1   = relu(ef @ w1 + b1)    -> WMMA, K padded 13->32, bias via C
//   layer 2:  Wout = (w2^T) @ h1^T + b2    -> WMMA, weights (A) resident in VGPRs
//
// Transposed formulation for layer 2 puts the edge index in the N dimension,
// so each lane's D fragment holds 8 *contiguous output channels* of one edge
// -> single coalesced non-temporal global_store_b128 per tile per lane
// (Wf is 410 MB streaming, > L2; NT keeps hot graph state resident).
//
// Per wave: 16 edges per tile, 10 tiles (grid-stride, uniform trip count so
// EXEC is all-ones at every WMMA).
// ---------------------------------------------------------------------------
__global__ __launch_bounds__(256) void fnet_kernel(
    const float* __restrict__ ef,   // [E,13]
    const float* __restrict__ w1,   // [13,32]
    const float* __restrict__ b1,   // [32]
    const float* __restrict__ w2,   // [32,256]
    const float* __restrict__ b2,   // [256]
    _Float16* __restrict__ Wout)    // [E,256] f16
{
    __shared__ alignas(16) _Float16 s_w2h[HID * 256];  // 16 KB, w2 as f16
    __shared__ alignas(16) _Float16 s_w1h[32 * 32];    // 2 KB, w1 padded K->32
    __shared__ alignas(32) float    s_b2f[256];
    __shared__ float                s_b1f[HID];
    __shared__ alignas(16) _Float16 s_h1[8 * 512];     // 8 KB, per-wave h1^T bounce

    const int tid = threadIdx.x;
    for (int i = tid; i < HID * 256; i += 256) s_w2h[i] = (_Float16)w2[i];
    for (int i = tid; i < 32 * 32;   i += 256) {
        const int k = i >> 5;
        s_w1h[i] = (k < FDIM) ? (_Float16)w1[i] : (_Float16)0.f;  // zero-pad K
    }
    for (int i = tid; i < 256; i += 256) s_b2f[i] = b2[i];
    for (int i = tid; i < HID; i += 256) s_b1f[i] = b1[i];
    __syncthreads();

    const int wave  = tid >> 5;
    const int lane  = tid & 31;
    const int ncol  = lane & 15;        // N index within a 16-wide tile
    const int g     = lane >> 4;        // lane group 0/1
    const int kb    = g * 8;            // A-fragment K base
    const int krow0 = g * 16;           // B-fragment K base
    const int wbase = wave * 512;       // per-wave h1 bounce region (halves)

    // ---- Resident A fragments for layer 2: A2[nt] = w2^T tile (16 c-rows x 32 h)
    //      A layout: lane row = lane%16 (= local c), halves t<8 -> K=kb+t,
    //      t>=8 -> K=kb+16+(t-8).
    v16h A2[16];
#pragma unroll
    for (int nt = 0; nt < 16; ++nt) {
#pragma unroll
        for (int t = 0; t < 16; ++t) {
            const int h = (t < 8) ? (kb + t) : (kb + 16 + (t - 8));
            A2[nt][t] = s_w2h[h * 256 + nt * 16 + ncol];
        }
    }
    // ---- Resident B fragments for layer 1: w1 padded [32 x 32], two N tiles.
    //      B layout: lane n = lane%16, half j<8 -> K = kb + j, j>=8 -> K = kb + 16 + (j-8).
    v16h Bw1[2];
#pragma unroll
    for (int t1 = 0; t1 < 2; ++t1) {
#pragma unroll
        for (int j = 0; j < 16; ++j)
            Bw1[t1][j] = s_w1h[((j < 8) ? (kb + j) : (kb + 16 + (j - 8))) * 32 + t1 * 16 + ncol];
    }

    const int waveId = blockIdx.x * 8 + wave;

#pragma unroll 1
    for (int it = 0; it < TPW; ++it) {
        const long tile = (long)waveId + (long)it * NWAVES;
        const long e0   = tile * 16;

        // ---- Layer-1 A fragment: ef rows, K padded to 32 with zeros.
        //      lanes 0-15 hold K=0..7 (halves 0-7) + K=16..23 (zero);
        //      lanes 16-31 hold K=8..15 (8..12 valid) + K=24..31 (zero).
        v16h ae;
#pragma unroll
        for (int t = 0; t < 16; ++t) ae[t] = (_Float16)0.f;
        {
            const int  row  = ncol;
            const long base = (e0 + row) * FDIM;
            if (g == 0) {
#pragma unroll
                for (int t = 0; t < 8; ++t)
                    ae[t] = (_Float16)__builtin_nontemporal_load(&ef[base + t]);
            } else {
#pragma unroll
                for (int t = 0; t < 5; ++t)
                    ae[t] = (_Float16)__builtin_nontemporal_load(&ef[base + 8 + t]);
            }
        }

        // ---- Layer 1: h1 = relu(ef @ w1 + b1); bias through the C operand.
        //      D: rows = edges (g*8+r), col = h_local = ncol.
        //      Write h1^T to per-wave LDS in [h][edge] layout (contiguous b128).
#pragma unroll
        for (int t1 = 0; t1 < 2; ++t1) {
            const float bb = s_b1f[t1 * 16 + ncol];
            v8f c1;
#pragma unroll
            for (int r = 0; r < 8; ++r) c1[r] = bb;
            v8f h = __builtin_amdgcn_wmma_f32_16x16x32_f16(
                        false, ae, false, Bw1[t1], (short)0, c1, false, false);
            v8h p;
#pragma unroll
            for (int r = 0; r < 8; ++r) p[r] = (_Float16)fmaxf(h[r], 0.0f);
            *(v8h*)&s_h1[wbase + (t1 * 16 + ncol) * 16 + g * 8] = p;
        }

        // ---- B fragment for layer 2: h1^T (32 h x 16 edges).
        //      lane: edge = ncol, half j<8 -> h = kb + j, j>=8 -> h = kb + 16 + (j-8).
        v16h bh;
#pragma unroll
        for (int j = 0; j < 16; ++j)
            bh[j] = s_h1[wbase + ((j < 8) ? (kb + j) : (kb + 16 + (j - 8))) * 16 + ncol];

        // ---- Layer 2: 16 tiles, D[c_local][edge]; bias b2 via C operand.
        //      Lane stores 8 contiguous channels of edge (e0+ncol) -> NT b128.
        _Float16* wout_row = Wout + (e0 + ncol) * 256;
#pragma unroll
        for (int nt = 0; nt < 16; ++nt) {
            const v8f c2 = *(const v8f*)&s_b2f[nt * 16 + g * 8];
            v8f d = __builtin_amdgcn_wmma_f32_16x16x32_f16(
                        false, A2[nt], false, bh, (short)0, c2, false, false);
            v8h p;
#pragma unroll
            for (int r = 0; r < 8; ++r) p[r] = (_Float16)d[r];
            __builtin_nontemporal_store(p, (v8h*)(wout_row + nt * 16 + g * 8));
        }
    }
}

// ---------------------------------------------------------------------------
// Per-edge message: m = hx[src] @ W_e (f16 filters), scatter-add to agg[dst].
// Batched GEMV with per-edge weights -> VALU; W streamed non-temporally
// (410 MB > L2), hx/agg stay cache-resident; scatter via global f32 atomics.
// ---------------------------------------------------------------------------
__global__ __launch_bounds__(256) void msg_kernel(
    const float* __restrict__ hx,     // [N,16]
    const _Float16* __restrict__ Wf,  // [E,256]
    const int* __restrict__ src,
    const int* __restrict__ dst,
    float* __restrict__ agg)          // [N,16]
{
    const long e = (long)blockIdx.x * 256 + threadIdx.x;
    if (e >= E_EDGES) return;
    const int s = src[e];
    const int d = dst[e];

    float x[NCF];
    const float4* hp = (const float4*)(hx + (long)s * NCF);
#pragma unroll
    for (int q = 0; q < 4; ++q) {
        float4 v = hp[q];
        x[q * 4 + 0] = v.x; x[q * 4 + 1] = v.y;
        x[q * 4 + 2] = v.z; x[q * 4 + 3] = v.w;
    }

    float m[NCF];
#pragma unroll
    for (int o = 0; o < NCF; ++o) m[o] = 0.0f;

    const v8h* wv = (const v8h*)(Wf + e * 256);
#pragma unroll
    for (int i = 0; i < NCF; ++i) {
        const float xi = x[i];
        v8h w0 = __builtin_nontemporal_load(&wv[2 * i]);
        v8h w1 = __builtin_nontemporal_load(&wv[2 * i + 1]);
#pragma unroll
        for (int o = 0; o < 8; ++o) {
            m[o]     += xi * (float)w0[o];
            m[8 + o] += xi * (float)w1[o];
        }
    }

    float* ap = agg + (long)d * NCF;
#pragma unroll
    for (int o = 0; o < NCF; ++o) unsafeAtomicAdd(&ap[o], m[o]);
}

// ---------------------------------------------------------------------------
// GRU cell per node: x = agg/deg; h' = GRU(x, h).  Thread-per-node VALU.
// ---------------------------------------------------------------------------
__device__ __forceinline__ float sigm(float t) { return 1.0f / (1.0f + __expf(-t)); }

__global__ __launch_bounds__(256) void gru_kernel(
    const float* __restrict__ agg,   // [N,16]
    const float* __restrict__ deg,   // [N]
    const float* __restrict__ hin,   // [N,16]
    const float* __restrict__ w_ih,  // [48,16]
    const float* __restrict__ w_hh,  // [48,16]
    const float* __restrict__ b_ih,  // [48]
    const float* __restrict__ b_hh,  // [48]
    float* __restrict__ hout)        // [N,16]
{
    const int n = blockIdx.x * 256 + threadIdx.x;
    if (n >= N_NODES) return;

    const float inv = 1.0f / fmaxf(deg[n], 1.0f);
    float x[NCF], h[NCF];
#pragma unroll
    for (int j = 0; j < NCF; ++j) {
        x[j] = agg[(long)n * NCF + j] * inv;
        h[j] = hin[(long)n * NCF + j];
    }

    float r[NCF], z[NCF];
#pragma unroll
    for (int k = 0; k < NCF; ++k) {
        float ir = b_ih[k], hr = b_hh[k];
#pragma unroll
        for (int j = 0; j < NCF; ++j) {
            ir += x[j] * w_ih[k * NCF + j];
            hr += h[j] * w_hh[k * NCF + j];
        }
        r[k] = sigm(ir + hr);
    }
#pragma unroll
    for (int k = 0; k < NCF; ++k) {
        float iz = b_ih[NCF + k], hz = b_hh[NCF + k];
#pragma unroll
        for (int j = 0; j < NCF; ++j) {
            iz += x[j] * w_ih[(NCF + k) * NCF + j];
            hz += h[j] * w_hh[(NCF + k) * NCF + j];
        }
        z[k] = sigm(iz + hz);
    }
#pragma unroll
    for (int k = 0; k < NCF; ++k) {
        float in_ = b_ih[2 * NCF + k], hn = b_hh[2 * NCF + k];
#pragma unroll
        for (int j = 0; j < NCF; ++j) {
            in_ += x[j] * w_ih[(2 * NCF + k) * NCF + j];
            hn  += h[j] * w_hh[(2 * NCF + k) * NCF + j];
        }
        const float nn = tanhf(in_ + r[k] * hn);
        hout[(long)n * NCF + k] = (1.0f - z[k]) * nn + z[k] * h[k];
    }
}

// ---------------------------------------------------------------------------
extern "C" void kernel_launch(void* const* d_in, const int* in_sizes, int n_in,
                              void* d_out, int out_size, void* d_ws, size_t ws_size,
                              hipStream_t stream) {
    (void)in_sizes; (void)n_in; (void)out_size; (void)ws_size;

    const float* hx    = (const float*)d_in[0];
    const float* ef    = (const float*)d_in[1];
    const int*   esrc  = (const int*)d_in[2];
    const int*   edst  = (const int*)d_in[3];
    const float* w1    = (const float*)d_in[4];
    const float* b1    = (const float*)d_in[5];
    const float* w2    = (const float*)d_in[6];
    const float* b2    = (const float*)d_in[7];
    const float* gwih  = (const float*)d_in[8];
    const float* gwhh  = (const float*)d_in[9];
    const float* gbih  = (const float*)d_in[10];
    const float* gbhh  = (const float*)d_in[11];
    float* out = (float*)d_out;

    // Workspace layout:
    //   Wf16 : E*256 f16        (409.6 MB)
    //   agg  : N*16 f32         (3.2 MB)
    //   deg  : N    f32         (0.2 MB)
    char* ws = (char*)d_ws;
    _Float16* Wf = (_Float16*)ws;
    float* agg = (float*)(ws + (size_t)E_EDGES * 256 * sizeof(_Float16));
    float* deg = agg + (size_t)N_NODES * NCF;

    const int eBlocks = (E_EDGES + 255) / 256;                 // 3125
    const int nBlocks = (N_NODES + 255) / 256;                 // 196
    const int aggN    = N_NODES * NCF;
    const int fillAll = aggN + N_NODES;                        // agg + deg contiguous
    const int fBlocks = (fillAll + 255) / 256;
    const int aBlocks = (aggN + 255) / 256;

    // deg (+ agg) init and degree count
    fill_zero_kernel<<<fBlocks, 256, 0, stream>>>(agg, fillAll);
    deg_kernel<<<eBlocks, 256, 0, stream>>>(edst, deg);

    // filter net once (weight sharing over repeats)
    fnet_kernel<<<FBLOCKS, 256, 0, stream>>>(ef, w1, b1, w2, b2, Wf);

    // repeat 1: hx from input, write d_out
    msg_kernel<<<eBlocks, 256, 0, stream>>>(hx, Wf, esrc, edst, agg);
    gru_kernel<<<nBlocks, 256, 0, stream>>>(agg, deg, hx, gwih, gwhh, gbih, gbhh, out);

    // repeat 2: hx from d_out, write d_out
    fill_zero_kernel<<<aBlocks, 256, 0, stream>>>(agg, aggN);
    msg_kernel<<<eBlocks, 256, 0, stream>>>(out, Wf, esrc, edst, agg);
    gru_kernel<<<nBlocks, 256, 0, stream>>>(agg, deg, out, gwih, gwhh, gbih, gbhh, out);
}
